// ChunkFormerEncoder_21474836480520
// MI455X (gfx1250) — matrix operand order repaired and checked by reading, not compiled, not run on hardware
//
#include <hip/hip_runtime.h>

// ---------------------------------------------------------------------------
// ChunkFormer encoder for MI455X (gfx1250, wave32, WMMA f16->f32)
// ---------------------------------------------------------------------------

typedef __attribute__((ext_vector_type(16))) _Float16 v16h;
typedef __attribute__((ext_vector_type(8)))  _Float16 v8h;
typedef __attribute__((ext_vector_type(8)))  float    v8f;
typedef __attribute__((ext_vector_type(4)))  float    v4f;

#define N_CHUNK 30
#define CSIZE   519
#define CSTEP   512
#define FEATD   80
#define T_IN    15367
#define H1 259
#define W1 39
#define H2 129
#define W2 19
#define H3 64
#define W3 9
#define CSB 256          // subsample conv channels
#define DM  256          // model dim
#define NH  4
#define DKH 64
#define FFD 2048
#define NL  12
#define KCN 15
#define CC  64           // chunk size (subsampled frames per chunk)
#define TOT (N_CHUNK*CC) // 1920 tokens
#define LWIN 256         // left(128)+chunk(64)+right(64)
#define PPOS 319         // rel-pos table length = left + 2c + right - 1
#define LEFTC 128

// ======================= generic batched WMMA GEMM =========================
// C[m,n] = act( sum_k (A[m,k]+aadd[k]) * B[n,k] + bias[n] ) * alpha  (+ res)
// A: f32 row-major (lda, contiguous K).  B: f16 row-major (ldb, contiguous K).
// Batched over grid.z with z -> (n_batch = z/Hb, h_batch = z%Hb) offsets.
// Wave tile: 32(M) x 64(N) (2 A-frags x 4 B-frags = 8 WMMAs / 32-K step).
// Block: 256 thr = 8 waves arranged 4(M) x 2(N) -> 128 x 128 block tile.
__global__ __launch_bounds__(256) void gemm_wmma_kernel(
    const float* __restrict__ A, const _Float16* __restrict__ B,
    const float* __restrict__ bias, const float* __restrict__ res,
    float* __restrict__ C, _Float16* __restrict__ C16,
    const float* __restrict__ aadd,
    int M, int Nn, int K, int lda, int ldb, int ldc,
    float alpha, int act, int Hb,
    long sAn, long sAh, long sBn, long sBh, long sCn, long sCh, long sAddh)
{
    int z  = blockIdx.z;
    int nb = z / Hb, hb = z - nb * Hb;
    const float*    Ab   = A + nb * sAn + hb * sAh;
    const _Float16* Bb   = B + nb * sBn + hb * sBh;
    long            coff = nb * sCn + hb * sCh;
    const float* aaddb = aadd ? (aadd + hb * sAddh) : nullptr;

    int wave = threadIdx.x >> 5;
    int wm   = wave & 3;        // wave position in M (4)
    int wn   = wave >> 2;       // wave position in N (2)
    int lane = threadIdx.x & 31;
    int half = lane >> 4;       // 0: lanes 0-15, 1: lanes 16-31
    int lidx = lane & 15;
    int m0 = blockIdx.y * 128 + wm * 32;
    int n0 = blockIdx.x * 128 + wn * 64;
    if (m0 >= M) return;

    // A fragment rows (ISA 16-bit A 16x32 layout: lane half selects K group)
    int arow0 = m0 + lidx;      if (arow0 >= M) arow0 = M - 1;
    int arow1 = m0 + 16 + lidx; if (arow1 >= M) arow1 = M - 1;
    const float* ap0 = Ab + (long)arow0 * lda + half * 8;
    const float* ap1 = Ab + (long)arow1 * lda + half * 8;

    // B fragment rows (output columns); per-lane-half K groups 8h..8h+7, 16+8h..16+8h+7
    const _Float16* bp[4];
#pragma unroll
    for (int j = 0; j < 4; j++) {
        int cn = n0 + j * 16 + lidx; if (cn >= Nn) cn = Nn - 1;
        bp[j] = Bb + (long)cn * ldb + half * 8;
    }

    v8f acc[2][4] = {};

    for (int k0 = 0; k0 < K; k0 += 32) {
        // ---- A fragments: f32 load -> f16 convert (optional per-K add) ----
        v16h a0, a1;
        if (aaddb) {
            const float* adk = aaddb + k0 + half * 8;
            const float* p0 = ap0 + k0;
            const float* p1 = ap1 + k0;
#pragma unroll
            for (int t = 0; t < 8; t++) {
                float lo = adk[t], hi = adk[16 + t];
                a0[t]     = (_Float16)(p0[t]      + lo);
                a0[8 + t] = (_Float16)(p0[16 + t] + hi);
                a1[t]     = (_Float16)(p1[t]      + lo);
                a1[8 + t] = (_Float16)(p1[16 + t] + hi);
            }
        } else {
            const float* p0 = ap0 + k0;
            const float* p1 = ap1 + k0;
            v4f f0 = *(const v4f*)(p0);
            v4f f1 = *(const v4f*)(p0 + 4);
            v4f f2 = *(const v4f*)(p0 + 16);
            v4f f3 = *(const v4f*)(p0 + 20);
            v4f g0 = *(const v4f*)(p1);
            v4f g1 = *(const v4f*)(p1 + 4);
            v4f g2 = *(const v4f*)(p1 + 16);
            v4f g3 = *(const v4f*)(p1 + 20);
#pragma unroll
            for (int t = 0; t < 4; t++) {
                a0[t]      = (_Float16)f0[t];
                a0[4 + t]  = (_Float16)f1[t];
                a0[8 + t]  = (_Float16)f2[t];
                a0[12 + t] = (_Float16)f3[t];
                a1[t]      = (_Float16)g0[t];
                a1[4 + t]  = (_Float16)g1[t];
                a1[8 + t]  = (_Float16)g2[t];
                a1[12 + t] = (_Float16)g3[t];
            }
        }
        // ---- 4 B fragments reused by both A fragments: 8 WMMAs ----
#pragma unroll
        for (int j = 0; j < 4; j++) {
            v8h blo = *(const v8h*)(bp[j] + k0);
            v8h bhi = *(const v8h*)(bp[j] + k0 + 16);
            v16h b = __builtin_shufflevector(blo, bhi, 0, 1, 2, 3, 4, 5, 6, 7, 8, 9, 10, 11, 12, 13, 14, 15);
            acc[0][j] = __builtin_amdgcn_wmma_f32_16x16x32_f16(
                            false, a0, false, b, (short)0, acc[0][j], false, false);
            acc[1][j] = __builtin_amdgcn_wmma_f32_16x16x32_f16(
                            false, a1, false, b, (short)0, acc[1][j], false, false);
        }
    }

    // ---- epilogue: C layout lane=column, VGPR r = row r (+8 upper half) ----
#pragma unroll
    for (int f = 0; f < 2; f++) {
        int rowB = m0 + f * 16 + (half << 3);
#pragma unroll
        for (int j = 0; j < 4; j++) {
            int n = n0 + j * 16 + lidx;
            if (n >= Nn) continue;
            float bv = bias ? bias[n] : 0.f;
#pragma unroll
            for (int r = 0; r < 8; r++) {
                int m = rowB + r;
                if (m >= M) continue;
                float v = acc[f][j][r] + bv;
                if (act == 1)      v = fmaxf(v, 0.f);            // ReLU
                else if (act == 2) v = v / (1.f + __expf(-v));   // SiLU
                v *= alpha;
                long ci = coff + (long)m * ldc + n;
                if (res) v += res[ci];
                C[ci] = v;
                if (C16) C16[ci] = (_Float16)v;
            }
        }
    }
}

// ============================ helper kernels ===============================
__global__ void f32_to_f16_kernel(const float* __restrict__ s,
                                  _Float16* __restrict__ d, size_t n) {
    size_t i = (size_t)blockIdx.x * blockDim.x + threadIdx.x;
    if (i < n) d[i] = (_Float16)s[i];
}

__global__ void unfold_kernel(const float* __restrict__ xs, float* __restrict__ xu) {
    int i = blockIdx.x * blockDim.x + threadIdx.x;
    if (i >= N_CHUNK * CSIZE * FEATD) return;
    int f = i % FEATD; int r = i / FEATD; int t = r % CSIZE; int n = r / CSIZE;
    int src = n * CSTEP + t;
    xu[i] = (src < T_IN) ? xs[(size_t)src * FEATD + f] : 0.f;
}

// 3x3 stride-2 VALID conv, 1 -> 256 channels, ReLU.  out spatial-major [sp][co]
__global__ void conv1_kernel(const float* __restrict__ in,
                             const float* __restrict__ w,
                             const float* __restrict__ b,
                             float* __restrict__ out) {
    int i = blockIdx.x * blockDim.x + threadIdx.x;
    if (i >= H1 * W1 * CSB) return;
    int co = i & 255; int sp = i >> 8; int x = sp % W1; int y = sp / W1;
    float acc = b[co];
#pragma unroll
    for (int ky = 0; ky < 3; ky++)
#pragma unroll
        for (int kx = 0; kx < 3; kx++)
            acc += in[(2 * y + ky) * FEATD + (2 * x + kx)] * w[co * 9 + ky * 3 + kx];
    out[(size_t)sp * CSB + co] = fmaxf(acc, 0.f);
}

// depthwise 3x3 stride-2 VALID conv (+bias), spatial-major layout
__global__ void dwconv2d_kernel(const float* __restrict__ in,
                                const float* __restrict__ w,
                                const float* __restrict__ b,
                                float* __restrict__ out,
                                int Hi, int Wi, int Ho, int Wo) {
    int i = blockIdx.x * blockDim.x + threadIdx.x;
    if (i >= Ho * Wo * CSB) return;
    int ch = i & 255; int sp = i >> 8; int x = sp % Wo; int y = sp / Wo;
    float acc = b[ch];
#pragma unroll
    for (int ky = 0; ky < 3; ky++)
#pragma unroll
        for (int kx = 0; kx < 3; kx++)
            acc += in[((2 * y + ky) * (size_t)Wi + (2 * x + kx)) * CSB + ch]
                   * w[ch * 9 + ky * 3 + kx];
    out[(size_t)sp * CSB + ch] = acc;
}

// (t, f=9, co) -> projection input row (n*64+t), feature co*9+f
__global__ void gather_proj_kernel(const float* __restrict__ p3,
                                   float* __restrict__ xproj, int n) {
    int i = blockIdx.x * blockDim.x + threadIdx.x;
    if (i >= CC * (CSB * 9)) return;
    int feat = i % (CSB * 9); int t = i / (CSB * 9);
    int co = feat / 9; int f = feat - co * 9;
    xproj[((size_t)(n * CC + t)) * (CSB * 9) + feat] = p3[(size_t)(t * 9 + f) * CSB + co];
}

__global__ void pe_kernel(float* __restrict__ pe) {
    int i = blockIdx.x * blockDim.x + threadIdx.x;
    if (i >= PPOS * (DM / 2)) return;
    int j = i % (DM / 2); int p = i / (DM / 2);
    float pos = (float)(LEFTC + CC - 1 - p);                    // 191 .. -127
    float freq = expf(-(2.f * j) * logf(10000.f) / (float)DM);
    float ang = pos * freq;
    pe[(size_t)p * DM + 2 * j]     = sinf(ang);
    pe[(size_t)p * DM + 2 * j + 1] = cosf(ang);
}

// row LayerNorm, D=256, one block per row (in-place safe)
__global__ __launch_bounds__(256) void ln_kernel(const float* __restrict__ x,
                                                 const float* __restrict__ s,
                                                 const float* __restrict__ b,
                                                 float* __restrict__ y) {
    __shared__ float red[256];
    int row = blockIdx.x, t = threadIdx.x;
    float v = x[(size_t)row * DM + t];
    red[t] = v; __syncthreads();
    for (int o = 128; o > 0; o >>= 1) { if (t < o) red[t] += red[t + o]; __syncthreads(); }
    float mean = red[0] / DM; __syncthreads();
    float d = v - mean;
    red[t] = d * d; __syncthreads();
    for (int o = 128; o > 0; o >>= 1) { if (t < o) red[t] += red[t + o]; __syncthreads(); }
    float r = rsqrtf(red[0] / DM + 1e-5f);
    y[(size_t)row * DM + t] = d * r * s[t] + b[t];
}

// gather K/V streaming windows; emit f16 K (row-major over D) and
// transposed V (per head: DK rows x Lw contiguous) for WMMA B operands
__global__ void window_gather_kernel(const float* __restrict__ k,
                                     const float* __restrict__ v,
                                     _Float16* __restrict__ kw16,
                                     _Float16* __restrict__ vwT16) {
    int i = blockIdx.x * blockDim.x + threadIdx.x;
    if (i >= N_CHUNK * LWIN * DM) return;
    int d = i % DM; int r = i / DM; int l = r % LWIN; int n = r / LWIN;
    int src = n * CC - LEFTC + l;
    bool ok = (src >= 0) && (src < TOT);
    float kv = ok ? k[(size_t)src * DM + d] : 0.f;
    float vv = ok ? v[(size_t)src * DM + d] : 0.f;
    kw16[((size_t)(n * LWIN + l)) * DM + d] = (_Float16)kv;
    int h = d >> 6, dk = d & 63;
    vwT16[(((size_t)(n * NH + h)) * DKH + dk) * LWIN + l] = (_Float16)vv;
}

// rel-shift gather of bd, combine with ac, mask, softmax, re-mask
__global__ __launch_bounds__(256) void attn_softmax_kernel(
    const float* __restrict__ ac, const float* __restrict__ bdf,
    float* __restrict__ att) {
    __shared__ float red[256];
    int bi = blockIdx.x;                      // bi = (n*NH+h)*CC + q
    int q = bi & 63;
    int n = bi / (NH * CC);
    int kk = threadIdx.x;
    int g = n * CC - LEFTC + kk;
    bool valid = (g >= 0) && (g < TOT);
    float s = ac[(size_t)bi * LWIN + kk]
            + bdf[(size_t)bi * PPOS + (CC - 1 + kk - q)];
    s = valid ? s * 0.125f : -1e4f;           // 1/sqrt(DK)=0.125
    red[kk] = s; __syncthreads();
    for (int o = 128; o > 0; o >>= 1) { if (kk < o) red[kk] = fmaxf(red[kk], red[kk + o]); __syncthreads(); }
    float mx = red[0]; __syncthreads();
    float e = __expf(s - mx);
    red[kk] = e; __syncthreads();
    for (int o = 128; o > 0; o >>= 1) { if (kk < o) red[kk] += red[kk + o]; __syncthreads(); }
    float inv = 1.f / red[0];
    att[(size_t)bi * LWIN + kk] = valid ? e * inv : 0.f;
}

__global__ void glu_kernel(const float* __restrict__ y2, float* __restrict__ g) {
    int i = blockIdx.x * blockDim.x + threadIdx.x;
    if (i >= TOT * DM) return;
    int ch = i & 255; int t = i >> 8;
    float a1 = y2[(size_t)t * (2 * DM) + ch];
    float a2 = y2[(size_t)t * (2 * DM) + DM + ch];
    g[i] = a1 / (1.f + __expf(-a2));
}

// depthwise conv1d k=15 over full 1920-frame sequence + BN(inference) + swish
__global__ void dwconv1d_kernel(const float* __restrict__ in,
                                const float* __restrict__ w,
                                const float* __restrict__ b,
                                const float* __restrict__ bg,
                                const float* __restrict__ bb,
                                float* __restrict__ out) {
    int i = blockIdx.x * blockDim.x + threadIdx.x;
    if (i >= TOT * DM) return;
    int ch = i & 255; int t = i >> 8;
    float acc = b[ch];
#pragma unroll
    for (int j = 0; j < KCN; j++) {
        int tt = t - (KCN / 2) + j;
        if ((unsigned)tt < (unsigned)TOT) acc += in[(size_t)tt * DM + ch] * w[ch * KCN + j];
    }
    acc = acc * 0.9999950000374997f * bg[ch] + bb[ch];   // /sqrt(1+1e-5) * g + b
    out[i] = acc / (1.f + __expf(-acc));                 // swish
}

// ============================== host side ==================================
static void launch_gemm(hipStream_t st, const float* A, const _Float16* B,
                        const float* bias, const float* res, float* C,
                        _Float16* C16, const float* aadd,
                        int M, int N, int K, int lda, int ldb, int ldc,
                        float alpha, int act, int batch, int Hb,
                        long sAn, long sAh, long sBn, long sBh,
                        long sCn, long sCh, long sAddh) {
    dim3 grid((unsigned)((N + 127) / 128), (unsigned)((M + 127) / 128), (unsigned)batch);
    gemm_wmma_kernel<<<grid, dim3(256), 0, st>>>(A, B, bias, res, C, C16, aadd,
        M, N, K, lda, ldb, ldc, alpha, act, Hb, sAn, sAh, sBn, sBh, sCn, sCh, sAddh);
}
static void launch_gemm_s(hipStream_t st, const float* A, const _Float16* B,
                          const float* bias, const float* res, float* C,
                          _Float16* C16, int M, int N, int K,
                          int lda, int ldb, int ldc, float alpha, int act) {
    launch_gemm(st, A, B, bias, res, C, C16, nullptr, M, N, K, lda, ldb, ldc,
                alpha, act, 1, 1, 0, 0, 0, 0, 0, 0, 0);
}

extern "C" void kernel_launch(void* const* d_in, const int* in_sizes, int n_in,
                              void* d_out, int out_size, void* d_ws, size_t ws_size,
                              hipStream_t stream) {
    (void)in_sizes; (void)n_in; (void)out_size; (void)ws_size;
    const float* xs      = (const float*)d_in[0];
    const float* conv1_w = (const float*)d_in[1];
    const float* conv1_b = (const float*)d_in[2];
    const float* dw2_w   = (const float*)d_in[3];
    const float* dw2_b   = (const float*)d_in[4];
    const float* pw2_w   = (const float*)d_in[5];
    const float* pw2_b   = (const float*)d_in[6];
    const float* dw3_w   = (const float*)d_in[7];
    const float* dw3_b   = (const float*)d_in[8];
    const float* pw3_w   = (const float*)d_in[9];
    const float* pw3_b   = (const float*)d_in[10];
    const float* proj_w  = (const float*)d_in[11];
    const float* proj_b  = (const float*)d_in[12];
    const float* ln1_s   = (const float*)d_in[13];
    const float* ln1_b   = (const float*)d_in[14];
    const float* ffm_w1  = (const float*)d_in[15];
    const float* ffm_b1  = (const float*)d_in[16];
    const float* ffm_w2  = (const float*)d_in[17];
    const float* ffm_b2  = (const float*)d_in[18];
    const float* ln2_s   = (const float*)d_in[19];
    const float* ln2_b   = (const float*)d_in[20];
    const float* q_w     = (const float*)d_in[21];
    const float* q_b     = (const float*)d_in[22];
    const float* k_w     = (const float*)d_in[23];
    const float* k_b     = (const float*)d_in[24];
    const float* v_w     = (const float*)d_in[25];
    const float* v_b     = (const float*)d_in[26];
    const float* pos_w   = (const float*)d_in[27];
    const float* out_w   = (const float*)d_in[28];
    const float* out_b   = (const float*)d_in[29];
    const float* bias_u  = (const float*)d_in[30];
    const float* bias_v  = (const float*)d_in[31];
    const float* ln3_s   = (const float*)d_in[32];
    const float* ln3_b   = (const float*)d_in[33];
    const float* cpw1_w  = (const float*)d_in[34];
    const float* cpw1_b  = (const float*)d_in[35];
    const float* cdw_w   = (const float*)d_in[36];
    const float* cdw_b   = (const float*)d_in[37];
    const float* bn_g    = (const float*)d_in[38];
    const float* bn_b    = (const float*)d_in[39];
    const float* cpw2_w  = (const float*)d_in[40];
    const float* cpw2_b  = (const float*)d_in[41];
    const float* ln4_s   = (const float*)d_in[42];
    const float* ln4_b   = (const float*)d_in[43];
    const float* ff_w1   = (const float*)d_in[44];
    const float* ff_b1   = (const float*)d_in[45];
    const float* ff_w2   = (const float*)d_in[46];
    const float* ff_b2   = (const float*)d_in[47];
    const float* ln5_s   = (const float*)d_in[48];
    const float* ln5_b   = (const float*)d_in[49];
    const float* aft_s   = (const float*)d_in[50];
    const float* aft_b   = (const float*)d_in[51];

    // ---- workspace bump allocator ----
    char* base = (char*)d_ws; size_t off = 0;
    auto alloc = [&](size_t bytes) -> char* {
        char* p = base + off; off = (off + bytes + 255) & ~(size_t)255; return p;
    };
    auto cvt = [&](const float* s, _Float16* dst, size_t n) {
        f32_to_f16_kernel<<<(unsigned)((n + 255) / 256), 256, 0, stream>>>(s, dst, n);
    };

    // f16 weight pool (B operands)
    _Float16* ffm1h = (_Float16*)alloc(2ull * NL * FFD * DM);
    _Float16* ffm2h = (_Float16*)alloc(2ull * NL * DM * FFD);
    _Float16* qwh   = (_Float16*)alloc(2ull * NL * DM * DM);
    _Float16* kwh   = (_Float16*)alloc(2ull * NL * DM * DM);
    _Float16* vwh   = (_Float16*)alloc(2ull * NL * DM * DM);
    _Float16* poswh = (_Float16*)alloc(2ull * NL * DM * DM);
    _Float16* outwh = (_Float16*)alloc(2ull * NL * DM * DM);
    _Float16* cp1h  = (_Float16*)alloc(2ull * NL * 2 * DM * DM);
    _Float16* cp2h  = (_Float16*)alloc(2ull * NL * DM * DM);
    _Float16* ff1h  = (_Float16*)alloc(2ull * NL * FFD * DM);
    _Float16* ff2h  = (_Float16*)alloc(2ull * NL * DM * FFD);
    _Float16* pw2h  = (_Float16*)alloc(2ull * CSB * CSB);
    _Float16* pw3h  = (_Float16*)alloc(2ull * CSB * CSB);
    _Float16* projh = (_Float16*)alloc(2ull * DM * (CSB * 9));

    // f32 activation buffers
    float* x     = (float*)alloc(4ull * TOT * DM);
    float* xn    = (float*)alloc(4ull * TOT * DM);
    float* ffh   = (float*)alloc(4ull * TOT * FFD);
    float* qb    = (float*)alloc(4ull * TOT * DM);
    float* kb    = (float*)alloc(4ull * TOT * DM);
    float* vb    = (float*)alloc(4ull * TOT * DM);
    _Float16* kw16  = (_Float16*)alloc(2ull * N_CHUNK * LWIN * DM);
    _Float16* vwT16 = (_Float16*)alloc(2ull * N_CHUNK * NH * DKH * LWIN);
    float* acb   = (float*)alloc(4ull * N_CHUNK * NH * CC * LWIN);
    float* bdf   = (float*)alloc(4ull * N_CHUNK * NH * CC * PPOS);
    float* attb  = (float*)alloc(4ull * N_CHUNK * NH * CC * LWIN);
    float* ao    = (float*)alloc(4ull * TOT * DM);
    float* y2    = (float*)alloc(4ull * TOT * 2 * DM);
    float* glub  = (float*)alloc(4ull * TOT * DM);
    float* dwo   = (float*)alloc(4ull * TOT * DM);
    float* peb   = (float*)alloc(4ull * PPOS * DM);
    float* ppf   = (float*)alloc(4ull * PPOS * DM);
    _Float16* pp16 = (_Float16*)alloc(2ull * PPOS * DM);
    float* xproj = (float*)alloc(4ull * TOT * (CSB * 9));
    float* xunf  = (float*)alloc(4ull * N_CHUNK * CSIZE * FEATD);
    float* c1buf = (float*)alloc(4ull * H1 * W1 * CSB);
    float* d2buf = (float*)alloc(4ull * H2 * W2 * CSB);
    float* p2buf = (float*)alloc(4ull * H2 * W2 * CSB);
    float* d3buf = (float*)alloc(4ull * H3 * W3 * CSB);
    float* p3buf = (float*)alloc(4ull * H3 * W3 * CSB);

    // ---- weight conversion (deterministic, re-done every call) ----
    cvt(ffm_w1, ffm1h, (size_t)NL * FFD * DM);
    cvt(ffm_w2, ffm2h, (size_t)NL * DM * FFD);
    cvt(q_w,   qwh,   (size_t)NL * DM * DM);
    cvt(k_w,   kwh,   (size_t)NL * DM * DM);
    cvt(v_w,   vwh,   (size_t)NL * DM * DM);
    cvt(pos_w, poswh, (size_t)NL * DM * DM);
    cvt(out_w, outwh, (size_t)NL * DM * DM);
    cvt(cpw1_w, cp1h, (size_t)NL * 2 * DM * DM);
    cvt(cpw2_w, cp2h, (size_t)NL * DM * DM);
    cvt(ff_w1, ff1h,  (size_t)NL * FFD * DM);
    cvt(ff_w2, ff2h,  (size_t)NL * DM * FFD);
    cvt(pw2_w, pw2h,  (size_t)CSB * CSB);
    cvt(pw3_w, pw3h,  (size_t)CSB * CSB);
    cvt(proj_w, projh, (size_t)DM * (CSB * 9));

    // ---- front-end: unfold + conv subsampling x8, chunk by chunk ----
    unfold_kernel<<<(N_CHUNK * CSIZE * FEATD + 255) / 256, 256, 0, stream>>>(xs, xunf);
    pe_kernel<<<(PPOS * (DM / 2) + 255) / 256, 256, 0, stream>>>(peb);

    for (int n = 0; n < N_CHUNK; n++) {
        const float* xc = xunf + (size_t)n * CSIZE * FEATD;
        conv1_kernel<<<(H1 * W1 * CSB + 255) / 256, 256, 0, stream>>>(xc, conv1_w, conv1_b, c1buf);
        dwconv2d_kernel<<<(H2 * W2 * CSB + 255) / 256, 256, 0, stream>>>(c1buf, dw2_w, dw2_b, d2buf, H1, W1, H2, W2);
        launch_gemm_s(stream, d2buf, pw2h, pw2_b, nullptr, p2buf, nullptr,
                      H2 * W2, CSB, CSB, CSB, CSB, CSB, 1.f, 1 /*relu*/);
        dwconv2d_kernel<<<(H3 * W3 * CSB + 255) / 256, 256, 0, stream>>>(p2buf, dw3_w, dw3_b, d3buf, H2, W2, H3, W3);
        launch_gemm_s(stream, d3buf, pw3h, pw3_b, nullptr, p3buf, nullptr,
                      H3 * W3, CSB, CSB, CSB, CSB, CSB, 1.f, 1 /*relu*/);
        gather_proj_kernel<<<(CC * CSB * 9 + 255) / 256, 256, 0, stream>>>(p3buf, xproj, n);
    }
    // input projection + xscale sqrt(D)=16
    launch_gemm_s(stream, xproj, projh, proj_b, nullptr, x, nullptr,
                  TOT, DM, CSB * 9, CSB * 9, CSB * 9, DM, 16.f, 0);

    // ---- 12 conformer blocks ----
    for (int i = 0; i < NL; i++) {
        // macaron FFN (0.5)
        ln_kernel<<<TOT, 256, 0, stream>>>(x, ln1_s + i * DM, ln1_b + i * DM, xn);
        launch_gemm_s(stream, xn, ffm1h + (size_t)i * FFD * DM, ffm_b1 + i * FFD,
                      nullptr, ffh, nullptr, TOT, FFD, DM, DM, DM, FFD, 1.f, 2 /*silu*/);
        launch_gemm_s(stream, ffh, ffm2h + (size_t)i * DM * FFD, ffm_b2 + i * DM,
                      x, x, nullptr, TOT, DM, FFD, FFD, FFD, DM, 0.5f, 0);

        // rel-pos chunked self-attention
        ln_kernel<<<TOT, 256, 0, stream>>>(x, ln2_s + i * DM, ln2_b + i * DM, xn);
        launch_gemm_s(stream, xn, qwh + (size_t)i * DM * DM, q_b + i * DM, nullptr, qb, nullptr, TOT, DM, DM, DM, DM, DM, 1.f, 0);
        launch_gemm_s(stream, xn, kwh + (size_t)i * DM * DM, k_b + i * DM, nullptr, kb, nullptr, TOT, DM, DM, DM, DM, DM, 1.f, 0);
        launch_gemm_s(stream, xn, vwh + (size_t)i * DM * DM, v_b + i * DM, nullptr, vb, nullptr, TOT, DM, DM, DM, DM, DM, 1.f, 0);
        window_gather_kernel<<<(N_CHUNK * LWIN * DM + 255) / 256, 256, 0, stream>>>(kb, vb, kw16, vwT16);
        launch_gemm_s(stream, peb, poswh + (size_t)i * DM * DM, nullptr, nullptr,
                      ppf, pp16, PPOS, DM, DM, DM, DM, DM, 1.f, 0);
        // ac = (q + bias_u) K^T   (batched over (chunk, head))
        launch_gemm(stream, qb, kw16, nullptr, nullptr, acb, nullptr,
                    bias_u + (size_t)i * NH * DKH,
                    CC, LWIN, DKH, DM, DM, LWIN, 1.f, 0, N_CHUNK * NH, NH,
                    (long)CC * DM, DKH, (long)LWIN * DM, DKH,
                    (long)NH * CC * LWIN, (long)CC * LWIN, DKH);
        // bd_full = (q + bias_v) pproj^T
        launch_gemm(stream, qb, pp16, nullptr, nullptr, bdf, nullptr,
                    bias_v + (size_t)i * NH * DKH,
                    CC, PPOS, DKH, DM, DM, PPOS, 1.f, 0, N_CHUNK * NH, NH,
                    (long)CC * DM, DKH, 0, DKH,
                    (long)NH * CC * PPOS, (long)CC * PPOS, DKH);
        attn_softmax_kernel<<<N_CHUNK * NH * CC, 256, 0, stream>>>(acb, bdf, attb);
        // ao = att V
        launch_gemm(stream, attb, vwT16, nullptr, nullptr, ao, nullptr, nullptr,
                    CC, DKH, LWIN, LWIN, LWIN, DM, 1.f, 0, N_CHUNK * NH, NH,
                    (long)NH * CC * LWIN, (long)CC * LWIN,
                    (long)NH * DKH * LWIN, (long)DKH * LWIN,
                    (long)CC * DM, DKH, 0);
        launch_gemm_s(stream, ao, outwh + (size_t)i * DM * DM, out_b + i * DM,
                      x, x, nullptr, TOT, DM, DM, DM, DM, DM, 1.f, 0);

        // conv module
        ln_kernel<<<TOT, 256, 0, stream>>>(x, ln3_s + i * DM, ln3_b + i * DM, xn);
        launch_gemm_s(stream, xn, cp1h + (size_t)i * 2 * DM * DM, cpw1_b + i * 2 * DM,
                      nullptr, y2, nullptr, TOT, 2 * DM, DM, DM, DM, 2 * DM, 1.f, 0);
        glu_kernel<<<(TOT * DM + 255) / 256, 256, 0, stream>>>(y2, glub);
        dwconv1d_kernel<<<(TOT * DM + 255) / 256, 256, 0, stream>>>(
            glub, cdw_w + (size_t)i * DM * KCN, cdw_b + i * DM,
            bn_g + i * DM, bn_b + i * DM, dwo);
        launch_gemm_s(stream, dwo, cp2h + (size_t)i * DM * DM, cpw2_b + i * DM,
                      x, x, nullptr, TOT, DM, DM, DM, DM, DM, 1.f, 0);

        // final FFN (0.5) + per-layer norm
        ln_kernel<<<TOT, 256, 0, stream>>>(x, ln4_s + i * DM, ln4_b + i * DM, xn);
        launch_gemm_s(stream, xn, ff1h + (size_t)i * FFD * DM, ff_b1 + i * FFD,
                      nullptr, ffh, nullptr, TOT, FFD, DM, DM, DM, FFD, 1.f, 2 /*silu*/);
        launch_gemm_s(stream, ffh, ff2h + (size_t)i * DM * FFD, ff_b2 + i * DM,
                      x, x, nullptr, TOT, DM, FFD, FFD, FFD, DM, 0.5f, 0);
        ln_kernel<<<TOT, 256, 0, stream>>>(x, ln5_s + i * DM, ln5_b + i * DM, x);
    }

    // after_norm -> output (30, 64, 256) f32
    ln_kernel<<<TOT, 256, 0, stream>>>(x, aft_s, aft_b, (float*)d_out);
}
